// Encoder_2525440770500
// MI455X (gfx1250) — matrix operand order repaired and checked by reading, not compiled, not run on hardware
//
#include <hip/hip_runtime.h>
#include <cstdint>
#include <cstddef>

// ---------------- problem constants ----------------
#define BB 2
#define SS 1024
#define DD 1024
#define HH 16
#define DKK 64
#define LL 4
#define FFF 2048
#define BSS (BB * SS)

// ---------------- types ----------------
typedef __attribute__((ext_vector_type(16))) __bf16 v16bf;
typedef __attribute__((ext_vector_type(8)))  float  v8f;

struct __align__(16) U4 { unsigned int x, y, z, w; };

union FragBF {
    U4     u[2];
    v16bf  v;
};

union Pack8 {
    unsigned short us[8];
    U4             u;
};

__device__ __forceinline__ unsigned short f2bf(float f) {
    unsigned int u = __builtin_bit_cast(unsigned int, f);
    u += 0x7FFFu + ((u >> 16) & 1u);   // round-to-nearest-even
    return (unsigned short)(u >> 16);
}

__device__ __forceinline__ v8f wmma_bf16(const FragBF& a, const FragBF& b, v8f c) {
    return __builtin_amdgcn_wmma_f32_16x16x32_bf16(
        false, a.v, false, b.v, (short)0, c, false, false);
}

// 16-lane reductions via DPP (quad_perm xor1, xor2, row_half_mirror, row_mirror)
// Stays inside each 16-lane half of the wave: exactly the WMMA C-layout row span.
#define DPP_MAXSTEP(x, ctrl)                                                  \
    x = fmaxf(x, __builtin_bit_cast(float, __builtin_amdgcn_mov_dpp(          \
                     __builtin_bit_cast(int, x), ctrl, 0xF, 0xF, true)))
#define DPP_SUMSTEP(x, ctrl)                                                  \
    x = x + __builtin_bit_cast(float, __builtin_amdgcn_mov_dpp(               \
                __builtin_bit_cast(int, x), ctrl, 0xF, 0xF, true))

__device__ __forceinline__ float red_max16(float x) {
    DPP_MAXSTEP(x, 0xB1);   // quad_perm(1,0,3,2)  xor1
    DPP_MAXSTEP(x, 0x4E);   // quad_perm(2,3,0,1)  xor2
    DPP_MAXSTEP(x, 0x141);  // row_half_mirror     (8-group swap)
    DPP_MAXSTEP(x, 0x140);  // row_mirror          (16-group swap)
    return x;
}
__device__ __forceinline__ float red_sum16(float x) {
    DPP_SUMSTEP(x, 0xB1);
    DPP_SUMSTEP(x, 0x4E);
    DPP_SUMSTEP(x, 0x141);
    DPP_SUMSTEP(x, 0x140);
    return x;
}

// ---------------- fp32 [K,N] -> bf16 [N,K] tiled transpose-convert ----------
__global__ __launch_bounds__(256) void convert_transpose_kernel(
    const float* __restrict__ in,        // [L][K][N]
    unsigned short* __restrict__ out,    // [L][N][K]
    int K, int N) {
    __shared__ unsigned short tile[32][33];
    const size_t layer = (size_t)blockIdx.z * K * N;
    const int n0 = blockIdx.x * 32;
    const int k0 = blockIdx.y * 32;
    const int tx = threadIdx.x & 31;
    const int ty = threadIdx.x >> 5;    // 0..7
#pragma unroll
    for (int i = 0; i < 4; i++) {
        int kl = ty + i * 8;
        tile[kl][tx] = f2bf(in[layer + (size_t)(k0 + kl) * N + n0 + tx]);
    }
    __syncthreads();
#pragma unroll
    for (int i = 0; i < 4; i++) {
        int nl = ty + i * 8;
        out[layer + (size_t)(n0 + nl) * K + k0 + tx] = tile[tx][nl];
    }
}

// ---------------- embedding + positional encoding ----------------
// Faithful to reference bug: pe indexed by BATCH (pos = b), broadcast over seq.
__global__ void embed_pe_kernel(const int* __restrict__ tokens,
                                const float* __restrict__ emb,
                                float* __restrict__ xf,
                                unsigned short* __restrict__ xb) {
    int idx = blockIdx.x * 256 + threadIdx.x;          // [0, B*S*D)
    int d  = idx & (DD - 1);
    int bs = idx >> 10;                                // / D
    int b  = bs >> 10;                                 // / S
    int tok = tokens[bs];
    float v = emb[(size_t)tok * DD + d];
    float two_i = (float)(d & ~1);
    float div = __expf(two_i * (-9.210340371976184f / (float)DD)); // ln(1e4)
    float ang = (float)b * div;
    v += (d & 1) ? __cosf(ang) : __sinf(ang);
    xf[idx] = v;
    xb[idx] = f2bf(v);
}

// ---------------- tiled WMMA GEMM, double-buffered LDS ----------------
// C[M,N] = A[M,K] * Wt[N,K]^T + bias.  A,Wt bf16 (Wt is k-major).
// MIRROR: 0 = none, 1 = bf16 row-major mirror, 2 = bf16 per-head transposed
// ([B,H,DK,S], vectorized 16B stores).  WF32: write fp32 C.
template <bool RELU, int MIRROR, bool WF32>
__global__ __launch_bounds__(256) void gemm_wmma_kernel(
    const unsigned short* __restrict__ A,
    const unsigned short* __restrict__ Wt,
    const float* __restrict__ bias,
    float* __restrict__ Cf,
    unsigned short* __restrict__ Cb,
    int M, int N, int K) {
    __shared__ __align__(16) unsigned short As[2][128][40];
    __shared__ __align__(16) unsigned short Bs[2][128][40];

    const int tid  = threadIdx.x;
    const int wave = tid >> 5;
    const int lane = tid & 31;
    const int lm   = lane & 15;
    const int kg   = lane >> 4;
    const int m0   = blockIdx.y * 128;
    const int n0   = blockIdx.x * 128;
    const int mw   = (wave >> 1) * 32;
    const int nw   = (wave & 1) * 64;

    const int srow = tid >> 1;            // 128 staging rows, 2 threads/row
    const int sseg = (tid & 1) * 16;
    const unsigned short* aptr = A  + (size_t)(m0 + srow) * K + sseg;
    const unsigned short* bptr = Wt + (size_t)(n0 + srow) * K + sseg;

    v8f acc[2][4];
#pragma unroll
    for (int i = 0; i < 2; i++)
#pragma unroll
        for (int j = 0; j < 4; j++)
            acc[i][j] = (v8f){0.f, 0.f, 0.f, 0.f, 0.f, 0.f, 0.f, 0.f};

    // prologue: stage k-step 0 into buffer 0
    {
        U4 a0 = *(const U4*)aptr, a1 = *(const U4*)(aptr + 8);
        U4 b0 = *(const U4*)bptr, b1 = *(const U4*)(bptr + 8);
        *(U4*)&As[0][srow][sseg]     = a0;
        *(U4*)&As[0][srow][sseg + 8] = a1;
        *(U4*)&Bs[0][srow][sseg]     = b0;
        *(U4*)&Bs[0][srow][sseg + 8] = b1;
    }

    const int nsteps = K >> 5;
    for (int s = 0; s < nsteps; s++) {
        __syncthreads();
        const int cur = s & 1, nxt = cur ^ 1;
        U4 a0, a1, b0, b1;
        const bool more = (s + 1 < nsteps);
        if (more) {   // issue next-tile global loads early; overlap with WMMA
            const unsigned short* ap = aptr + (s + 1) * 32;
            const unsigned short* bp = bptr + (s + 1) * 32;
            a0 = *(const U4*)ap; a1 = *(const U4*)(ap + 8);
            b0 = *(const U4*)bp; b1 = *(const U4*)(bp + 8);
        }

        FragBF a[2];
#pragma unroll
        for (int mt = 0; mt < 2; mt++) {
            const unsigned short* r = &As[cur][mw + mt * 16 + lm][0];
            a[mt].u[0] = *(const U4*)(r + kg * 8);        // K = kg*8 + e
            a[mt].u[1] = *(const U4*)(r + 16 + kg * 8);   // K = 16 + kg*8 + e
        }
#pragma unroll
        for (int nt = 0; nt < 4; nt++) {
            FragBF b;
            const unsigned short* r = &Bs[cur][nw + nt * 16 + lm][0];
            b.u[0] = *(const U4*)(r + kg * 8);            // K = kg*8 + e
            b.u[1] = *(const U4*)(r + 16 + kg * 8);
#pragma unroll
            for (int mt = 0; mt < 2; mt++)
                acc[mt][nt] = wmma_bf16(a[mt], b, acc[mt][nt]);
        }

        if (more) {
            *(U4*)&As[nxt][srow][sseg]     = a0;
            *(U4*)&As[nxt][srow][sseg + 8] = a1;
            *(U4*)&Bs[nxt][srow][sseg]     = b0;
            *(U4*)&Bs[nxt][srow][sseg + 8] = b1;
        }
    }

    // epilogue: C layout -> col = lm, rows = kg*8 + r
#pragma unroll
    for (int mt = 0; mt < 2; mt++) {
#pragma unroll
        for (int nt = 0; nt < 4; nt++) {
            const int n = n0 + nw + nt * 16 + lm;
            const float bsv = bias[n];
            const int row0 = m0 + mw + mt * 16 + kg * 8;
            if (MIRROR == 2) {
                // [B,H,DK,S]: 8 consecutive s -> single 16B store
                const int hh = n >> 6, dk = n & 63;
                const int bb = row0 >> 10, sr = row0 & 1023;
                Pack8 pk;
#pragma unroll
                for (int r = 0; r < 8; r++) {
                    float v = acc[mt][nt][r] + bsv;
                    if (RELU) v = v > 0.f ? v : 0.f;
                    pk.us[r] = f2bf(v);
                    if (WF32) Cf[(size_t)(row0 + r) * N + n] = v;
                }
                *(U4*)(Cb + ((size_t)(bb * HH + hh) * DKK + dk) * SS + sr) = pk.u;
            } else {
#pragma unroll
                for (int r = 0; r < 8; r++) {
                    float v = acc[mt][nt][r] + bsv;
                    if (RELU) v = v > 0.f ? v : 0.f;
                    const size_t o = (size_t)(row0 + r) * N + n;
                    if (WF32) Cf[o] = v;
                    if (MIRROR == 1) Cb[o] = f2bf(v);
                }
            }
        }
    }
}

// ---------------- flash attention: barrier-free, pipelined ----------------
// grid = B*H*(S/64); block = 128 (4 waves); wave handles 16 query rows.
// Q,K row-major [B*S,D]; V pre-transposed [B,H,DK,S]. Output bf16 row-major.
__global__ __launch_bounds__(128) void attn_kernel(
    const unsigned short* __restrict__ Q,
    const unsigned short* __restrict__ Km,
    const unsigned short* __restrict__ Vt,
    unsigned short* __restrict__ Ob) {
    __shared__ __align__(16) unsigned short Pl[4][16][40];   // per-wave P tile

    int blk = blockIdx.x;
    const int qt = blk & 15;  blk >>= 4;
    const int h  = blk & 15;  blk >>= 4;
    const int b  = blk;
    const int tid  = threadIdx.x;
    const int wave = tid >> 5;
    const int lane = tid & 31;
    const int lm   = lane & 15;
    const int kg   = lane >> 4;
    const int q0   = qt * 64 + wave * 16;

    FragBF aq[2];
    {
        const unsigned short* qrow =
            Q + ((size_t)(b * SS + q0 + lm)) * DD + h * DKK;
        aq[0].u[0] = *(const U4*)(qrow + kg * 8);
        aq[0].u[1] = *(const U4*)(qrow + 16 + kg * 8);
        aq[1].u[0] = *(const U4*)(qrow + 32 + kg * 8);
        aq[1].u[1] = *(const U4*)(qrow + 48 + kg * 8);
    }

    const unsigned short* kbase = Km + (size_t)(b * SS) * DD + h * DKK;
    const unsigned short* vbase = Vt + (size_t)(b * HH + h) * DKK * SS;

    v8f acc[4];
#pragma unroll
    for (int i = 0; i < 4; i++)
        acc[i] = (v8f){0.f, 0.f, 0.f, 0.f, 0.f, 0.f, 0.f, 0.f};
    float rmax[8], rsum[8];
#pragma unroll
    for (int r = 0; r < 8; r++) { rmax[r] = -1e30f; rsum[r] = 0.f; }

    auto loadK = [&](FragBF (&f)[2][2], int kb) {
#pragma unroll
        for (int t = 0; t < 2; t++)
#pragma unroll
            for (int ks = 0; ks < 2; ks++) {
                const unsigned short* p =
                    kbase + (size_t)(kb + t * 16 + lm) * DD + ks * 32 + kg * 8;
                f[t][ks].u[0] = *(const U4*)p;
                f[t][ks].u[1] = *(const U4*)(p + 16);
            }
    };
    auto loadV = [&](FragBF (&f)[4], int kb) {
#pragma unroll
        for (int nt = 0; nt < 4; nt++) {
            const unsigned short* p =
                vbase + (size_t)(nt * 16 + lm) * SS + kb + kg * 8;
            f[nt].u[0] = *(const U4*)p;
            f[nt].u[1] = *(const U4*)(p + 16);
        }
    };

    FragBF fk[2][2], fv[4];
    loadK(fk, 0);
    loadV(fv, 0);

    for (int kb = 0; kb < SS; kb += 32) {
        FragBF nk[2][2], nv[4];
        if (kb + 32 < SS) { loadK(nk, kb + 32); loadV(nv, kb + 32); }

        v8f s0 = (v8f){0.f, 0.f, 0.f, 0.f, 0.f, 0.f, 0.f, 0.f};
        v8f s1 = (v8f){0.f, 0.f, 0.f, 0.f, 0.f, 0.f, 0.f, 0.f};
        s0 = wmma_bf16(aq[0], fk[0][0], s0);
        s0 = wmma_bf16(aq[1], fk[0][1], s0);
        s1 = wmma_bf16(aq[0], fk[1][0], s1);
        s1 = wmma_bf16(aq[1], fk[1][1], s1);

        // online softmax: DPP 16-lane reductions, no LDS round-trips
#pragma unroll
        for (int r = 0; r < 8; r++) {
            float v0 = s0[r] * 0.125f;   // 1/sqrt(64)
            float v1 = s1[r] * 0.125f;
            float bm = red_max16(fmaxf(v0, v1));
            float mn   = fmaxf(rmax[r], bm);
            float corr = __expf(rmax[r] - mn);
            float e0   = __expf(v0 - mn);
            float e1   = __expf(v1 - mn);
            float rs   = red_sum16(e0 + e1);
            rsum[r] = rsum[r] * corr + rs;
            rmax[r] = mn;
#pragma unroll
            for (int nt = 0; nt < 4; nt++) acc[nt][r] *= corr;
            Pl[wave][kg * 8 + r][lm]      = f2bf(e0);
            Pl[wave][kg * 8 + r][16 + lm] = f2bf(e1);
        }

        // P (C-layout) -> A-layout via per-wave LDS (same-wave DS is in-order)
        FragBF ap;
        {
            const unsigned short* prow = &Pl[wave][lm][0];
            ap.u[0] = *(const U4*)(prow + kg * 8);
            ap.u[1] = *(const U4*)(prow + 16 + kg * 8);
        }
#pragma unroll
        for (int nt = 0; nt < 4; nt++)
            acc[nt] = wmma_bf16(ap, fv[nt], acc[nt]);

        if (kb + 32 < SS) {
#pragma unroll
            for (int t = 0; t < 2; t++)
#pragma unroll
                for (int ks = 0; ks < 2; ks++) fk[t][ks] = nk[t][ks];
#pragma unroll
            for (int nt = 0; nt < 4; nt++) fv[nt] = nv[nt];
        }
    }

    // epilogue: normalize, store bf16 row-major
#pragma unroll
    for (int nt = 0; nt < 4; nt++) {
#pragma unroll
        for (int r = 0; r < 8; r++) {
            const int row = q0 + kg * 8 + r;
            const float v = acc[nt][r] / rsum[r];
            Ob[((size_t)(b * SS + row)) * DD + h * DKK + nt * 16 + lm] = f2bf(v);
        }
    }
}

// ---------------- residual add + LayerNorm ----------------
__global__ __launch_bounds__(256) void add_ln_kernel(
    const float* __restrict__ x, const float* __restrict__ rsd,
    const float* __restrict__ g, const float* __restrict__ be,
    float* __restrict__ yf, unsigned short* __restrict__ yb) {
    __shared__ float red[8];
    __shared__ float stat[2];
    const int row  = blockIdx.x;
    const int tid  = threadIdx.x;
    const int wave = tid >> 5;
    const int lane = tid & 31;
    const float* xr = x   + (size_t)row * DD;
    const float* rr = rsd + (size_t)row * DD;

    float v[4];
    float s = 0.f;
#pragma unroll
    for (int i = 0; i < 4; i++) {
        int c = tid + i * 256;
        v[i] = xr[c] + rr[c];
        s += v[i];
    }
#pragma unroll
    for (int m = 16; m >= 1; m >>= 1) s += __shfl_xor(s, m, 32);
    if (lane == 0) red[wave] = s;
    __syncthreads();
    if (tid == 0) {
        float t = 0.f;
        for (int i = 0; i < 8; i++) t += red[i];
        stat[0] = t * (1.f / DD);
    }
    __syncthreads();
    float mu = stat[0];
    float q = 0.f;
#pragma unroll
    for (int i = 0; i < 4; i++) { float d = v[i] - mu; q += d * d; }
#pragma unroll
    for (int m = 16; m >= 1; m >>= 1) q += __shfl_xor(q, m, 32);
    if (lane == 0) red[wave] = q;
    __syncthreads();
    if (tid == 0) {
        float t = 0.f;
        for (int i = 0; i < 8; i++) t += red[i];
        stat[1] = t * (1.f / DD);
    }
    __syncthreads();
    float inv = rsqrtf(stat[1] + 1e-5f);
#pragma unroll
    for (int i = 0; i < 4; i++) {
        int c = tid + i * 256;
        float y = (v[i] - mu) * inv * g[c] + be[c];
        size_t o = (size_t)row * DD + c;
        yf[o] = y;
        yb[o] = f2bf(y);
    }
}

// ---------------- host orchestration ----------------
extern "C" void kernel_launch(void* const* d_in, const int* in_sizes, int n_in,
                              void* d_out, int out_size, void* d_ws, size_t ws_size,
                              hipStream_t stream) {
    (void)in_sizes; (void)n_in; (void)out_size; (void)ws_size;

    const int*   tokens = (const int*)d_in[0];
    const float* emb    = (const float*)d_in[1];
    const float* Wq     = (const float*)d_in[2];
    const float* bq     = (const float*)d_in[3];
    const float* Wk     = (const float*)d_in[4];
    const float* bk     = (const float*)d_in[5];
    const float* Wv     = (const float*)d_in[6];
    const float* bv     = (const float*)d_in[7];
    const float* Wo     = (const float*)d_in[8];
    const float* bo     = (const float*)d_in[9];
    const float* W1     = (const float*)d_in[10];
    const float* b1     = (const float*)d_in[11];
    const float* W2     = (const float*)d_in[12];
    const float* b2     = (const float*)d_in[13];
    const float* gamma  = (const float*)d_in[14];
    const float* beta   = (const float*)d_in[15];

    char* wsp = (char*)d_ws;
    auto alloc = [&](size_t bytes) -> void* {
        void* p = (void*)wsp;
        wsp += (bytes + 255) & ~(size_t)255;
        return p;
    };

    const size_t NDD = (size_t)LL * DD * DD;
    const size_t NDF = (size_t)LL * DD * FFF;

    unsigned short* wq_t = (unsigned short*)alloc(NDD * 2);  // [L][D][D] n-major
    unsigned short* wk_t = (unsigned short*)alloc(NDD * 2);
    unsigned short* wv_t = (unsigned short*)alloc(NDD * 2);
    unsigned short* wo_t = (unsigned short*)alloc(NDD * 2);
    unsigned short* w1_t = (unsigned short*)alloc(NDF * 2);  // [L][FF][D]
    unsigned short* w2_t = (unsigned short*)alloc(NDF * 2);  // [L][D][FF]

    float*          xf = (float*)alloc((size_t)BSS * DD * 4);
    unsigned short* xb = (unsigned short*)alloc((size_t)BSS * DD * 2);
    unsigned short* qb = (unsigned short*)alloc((size_t)BSS * DD * 2);
    unsigned short* kb2 = (unsigned short*)alloc((size_t)BSS * DD * 2);
    unsigned short* vt = (unsigned short*)alloc((size_t)BSS * DD * 2);  // [B,H,DK,S]
    unsigned short* ob = (unsigned short*)alloc((size_t)BSS * DD * 2);
    float*          pf = (float*)alloc((size_t)BSS * DD * 4);
    unsigned short* hb = (unsigned short*)alloc((size_t)BSS * FFF * 2);

    // one-shot weight convert + transpose to [N][K] bf16 (L2-resident afterwards)
    {
        dim3 gdd(DD / 32, DD / 32, LL);
        convert_transpose_kernel<<<gdd, 256, 0, stream>>>(Wq, wq_t, DD, DD);
        convert_transpose_kernel<<<gdd, 256, 0, stream>>>(Wk, wk_t, DD, DD);
        convert_transpose_kernel<<<gdd, 256, 0, stream>>>(Wv, wv_t, DD, DD);
        convert_transpose_kernel<<<gdd, 256, 0, stream>>>(Wo, wo_t, DD, DD);
        dim3 g1(FFF / 32, DD / 32, LL);   // W1 [D][FF] -> [FF][D]
        convert_transpose_kernel<<<g1, 256, 0, stream>>>(W1, w1_t, DD, FFF);
        dim3 g2(DD / 32, FFF / 32, LL);   // W2 [FF][D] -> [D][FF]
        convert_transpose_kernel<<<g2, 256, 0, stream>>>(W2, w2_t, FFF, DD);
    }

    embed_pe_kernel<<<(BSS * DD) / 256, 256, 0, stream>>>(tokens, emb, xf, xb);

    dim3 gDD(DD / 128, BSS / 128);
    dim3 gFFg(FFF / 128, BSS / 128);

    for (int l = 0; l < LL; l++) {
        const unsigned short* wql = wq_t + (size_t)l * DD * DD;
        const unsigned short* wkl = wk_t + (size_t)l * DD * DD;
        const unsigned short* wvl = wv_t + (size_t)l * DD * DD;
        const unsigned short* wol = wo_t + (size_t)l * DD * DD;
        const unsigned short* w1l = w1_t + (size_t)l * DD * FFF;
        const unsigned short* w2l = w2_t + (size_t)l * FFF * DD;
        const float* bql = bq + (size_t)l * DD;
        const float* bkl = bk + (size_t)l * DD;
        const float* bvl = bv + (size_t)l * DD;
        const float* bol = bo + (size_t)l * DD;
        const float* b1l = b1 + (size_t)l * FFF;
        const float* b2l = b2 + (size_t)l * DD;
        const float* gl  = gamma + (size_t)l * DD;
        const float* bel = beta  + (size_t)l * DD;

        gemm_wmma_kernel<false, 1, false><<<gDD, 256, 0, stream>>>(
            xb, wql, bql, nullptr, qb, BSS, DD, DD);
        gemm_wmma_kernel<false, 1, false><<<gDD, 256, 0, stream>>>(
            xb, wkl, bkl, nullptr, kb2, BSS, DD, DD);
        gemm_wmma_kernel<false, 2, false><<<gDD, 256, 0, stream>>>(
            xb, wvl, bvl, nullptr, vt, BSS, DD, DD);

        attn_kernel<<<BB * HH * (SS / 64), 128, 0, stream>>>(qb, kb2, vt, ob);

        gemm_wmma_kernel<false, 0, true><<<gDD, 256, 0, stream>>>(
            ob, wol, bol, pf, nullptr, BSS, DD, DD);

        add_ln_kernel<<<BSS, 256, 0, stream>>>(xf, pf, gl, bel, xf, xb);

        gemm_wmma_kernel<true, 1, false><<<gFFg, 256, 0, stream>>>(
            xb, w1l, b1l, nullptr, hb, BSS, FFF, DD);
        gemm_wmma_kernel<false, 0, true><<<gDD, 256, 0, stream>>>(
            hb, w2l, b2l, pf, nullptr, BSS, DD, FFF);

        float* outT = (l == LL - 1) ? (float*)d_out : xf;
        add_ln_kernel<<<BSS, 256, 0, stream>>>(xf, pf, gl, bel, outT, xb);
    }
}
